// LSTM_60524679135330
// MI455X (gfx1250) — matrix operand order repaired and checked by reading, not compiled, not run on hardware
//
#include <hip/hip_runtime.h>

// Problem dims
#define B_  32
#define T_  100
#define V_  10000
#define E_  128
#define H_  500
#define HP_ 512          // padded hidden
#define NG_ 2048         // 4 gates * HP_

typedef __attribute__((ext_vector_type(16))) _Float16     v16h;
typedef __attribute__((ext_vector_type(8)))  float        v8f;
typedef __attribute__((ext_vector_type(8)))  unsigned int v8u;

__device__ __forceinline__ v8f wmma_f16(v16h a, v16h b, v8f c) {
  // D = A(16x32 f16) * B(32x16 f16) + C(16x16 f32)
  return __builtin_amdgcn_wmma_f32_16x16x32_f16(false, a, false, b, (short)0, c,
                                                false, false);
}

__device__ __forceinline__ float sigf(float x) { return 1.0f / (1.0f + expf(-x)); }

// Inverse of the ISA 7.12.2 16-bit A-operand layout: element (m16, kw) of a
// 16x32 tile lives at half index (lane*16 + e) with lane = m16 + 16*h.
// Forward map: lane half h -> kbase=8h; vgpr p holds K pairs 2(p%4)+16(p/4).
__device__ __forceinline__ int a_blob_idx(int m16, int kw) {
  int h = (kw >> 3) & 1;
  int p = ((kw >> 4) << 2) + ((kw & 7) >> 1);
  int e = 2 * p + (kw & 1);
  return (m16 + 16 * h) * 16 + e;
}

// mask ints (0/1) -> packed f16 A fragment: half(x) == x * 0x3C00 exactly.
// q0,q1 = 8 contiguous ints at k0; q2,q3 = 8 contiguous ints at k0+16.
__device__ __forceinline__ v16h pack_mask_a(int4 q0, int4 q1, int4 q2, int4 q3) {
  v8u u;
  u[0] = (unsigned)q0.x * 0x3C00u + (unsigned)q0.y * 0x3C000000u;
  u[1] = (unsigned)q0.z * 0x3C00u + (unsigned)q0.w * 0x3C000000u;
  u[2] = (unsigned)q1.x * 0x3C00u + (unsigned)q1.y * 0x3C000000u;
  u[3] = (unsigned)q1.z * 0x3C00u + (unsigned)q1.w * 0x3C000000u;
  u[4] = (unsigned)q2.x * 0x3C00u + (unsigned)q2.y * 0x3C000000u;
  u[5] = (unsigned)q2.z * 0x3C00u + (unsigned)q2.w * 0x3C000000u;
  u[6] = (unsigned)q3.x * 0x3C00u + (unsigned)q3.y * 0x3C000000u;
  u[7] = (unsigned)q3.z * 0x3C00u + (unsigned)q3.w * 0x3C000000u;
  union { v8u u; v16h h; } cvt;
  cvt.u = u;
  return cvt.h;
}

// ---------------------------------------------------------------------------
// Pack fp32 weight matrix [srcK][srcStride] -> f16 WMMA B-operand tiles.
// Gate remap: packed col = g*Jpad + j  <-  src col = g*J + j   (zero padded).
// Tile (32x16 = 512 halves): lane c (0..15) = col c, K = p<8 ? p : 8+p; lane 16+c: K=8+(same).
// ---------------------------------------------------------------------------
__global__ void pack_b_kernel(const float* __restrict__ src, int srcK, int srcStride,
                              int J, int Jpad, int numKt, int numNt,
                              _Float16* __restrict__ dst) {
  size_t total = (size_t)numKt * numNt * 512;
  size_t d = (size_t)blockIdx.x * blockDim.x + threadIdx.x;
  if (d >= total) return;
  int r = (int)(d & 511);
  size_t tile = d >> 9;
  int kt = (int)(tile % numKt);
  int nt = (int)(tile / numKt);
  int lane = r >> 4;
  int p = r & 15;
  int K = kt * 32 + ((lane >= 16) ? 8 : 0) + (p & 7) + ((p >> 3) << 4);
  int col = nt * 16 + (lane & 15);
  int g = col / Jpad;
  int j = col % Jpad;
  float v = 0.0f;
  if (K < srcK && j < J) v = src[(size_t)K * srcStride + g * J + j];
  dst[d] = (_Float16)v;
}

__global__ void pack_bias_kernel(const float* __restrict__ src, int G, int J, int Jpad,
                                 float* __restrict__ dst) {
  int i = blockIdx.x * blockDim.x + threadIdx.x;
  if (i >= G * Jpad) return;
  int g = i / Jpad, j = i % Jpad;
  dst[i] = (j < J) ? src[g * J + j] : 0.0f;
}

// ---------------------------------------------------------------------------
// e = mask @ emb via f16 WMMA.  One block per 16-row M-tile (rows b-major,
// r = b*T + t); 8 waves split K; fp32 LDS reduction; result stored as
// A-operand blobs over t-major rows (row' = t*B + b), 4 K-tiles of 512 halves
// per M-tile.  Main K loop guard-free; kk==312 handled by split 0.
// ---------------------------------------------------------------------------
__global__ void __launch_bounds__(256)
embed_gemm_kernel(const int* __restrict__ mask, const _Float16* __restrict__ Bpk,
                  _Float16* __restrict__ Eo, int numKt /*313*/) {
  __shared__ float red[8 * 2048];
  int mtile = blockIdx.x;                       // 0..199
  int split = threadIdx.x >> 5;                 // 0..7 (K split)
  int lane  = threadIdx.x & 31;
  int kbase = (lane & 16) ? 8 : 0;
  const int* rowp = mask + (size_t)(mtile * 16 + (lane & 15)) * V_;
  v8f acc[8] = {};
  int kLast = numKt - 1;                        // 312, the only guarded iter
  for (int kk = split; kk < kLast; kk += 8) {
    int k0 = kk * 32 + kbase;
    const int4* p0 = (const int4*)(rowp + k0);
    const int4* p1 = (const int4*)(rowp + k0 + 16);
    v16h a = pack_mask_a(p0[0], p0[1], p1[0], p1[1]);
#pragma unroll
    for (int nt = 0; nt < 8; ++nt) {
      const v16h b = *(const v16h*)(Bpk + (((size_t)nt * numKt + kk) << 9) + lane * 16);
      acc[nt] = wmma_f16(a, b, acc[nt]);
    }
  }
  if (split == 0) {                             // tail: K = 9984..9999 valid
    int k0 = kLast * 32 + kbase;
    const int4* p0 = (const int4*)(rowp + k0);
    int4 zz = {0, 0, 0, 0};
    v16h a = pack_mask_a(p0[0], p0[1], zz, zz);
#pragma unroll
    for (int nt = 0; nt < 8; ++nt) {
      const v16h b =
          *(const v16h*)(Bpk + (((size_t)nt * numKt + kLast) << 9) + lane * 16);
      acc[nt] = wmma_f16(a, b, acc[nt]);
    }
  }
#pragma unroll
  for (int nt = 0; nt < 8; ++nt)
    *(v8f*)(&red[split * 2048 + nt * 256 + lane * 8]) = acc[nt];
  __syncthreads();
  for (int f = threadIdx.x; f < 2048; f += 256) {
    float s = 0.0f;
#pragma unroll
    for (int sp = 0; sp < 8; ++sp) s += red[sp * 2048 + f];
    // decode blob element: f = nt*256 + ln*8 + v
    int nt = f >> 8, rem = f & 255, ln = rem >> 3, v = rem & 7;
    int m = ((ln >> 4) << 3) + v;
    int col = (nt << 4) + (ln & 15);
    int r = mtile * 16 + m;                     // b-major row
    int bb = r / T_, tt = r % T_;
    int rp = tt * B_ + bb;                      // t-major row'
    Eo[((size_t)(rp >> 4) * 4 + (col >> 5)) * 512 + a_blob_idx(rp & 15, col & 31)] =
        (_Float16)s;
  }
}

// ---------------------------------------------------------------------------
// Generic C = Apk(A-blob f16) @ Bpk + bias.  Wave: 16-row M-tile x 8 N-tiles.
// All operand fetches are single v16h (2 x b128) loads.
// Output: row-major fp32 (opt tanh) or z tile blobs (256 fp32, idx lane*8+v)
// stored as one v8f per tile for the LSTM scan accumulator seed.
// ---------------------------------------------------------------------------
__global__ void gemm_f16_kernel(const _Float16* __restrict__ Apk,
                                const _Float16* __restrict__ Bpk,
                                const float* __restrict__ bias,
                                float* __restrict__ C, int ldc,
                                int numMtiles, int numChunks, int numKt,
                                int act, int tileStore) {
  int w = blockIdx.x * 8 + (threadIdx.x >> 5);
  if (w >= numMtiles * numChunks) return;
  int lane = threadIdx.x & 31;
  int mtile = w % numMtiles;
  int chunk = w / numMtiles;
  const _Float16* Abase = Apk + ((size_t)mtile * numKt) * 512 + lane * 16;
  v8f acc[8] = {};
  for (int kk = 0; kk < numKt; ++kk) {
    v16h a = *(const v16h*)(Abase + (size_t)kk * 512);
#pragma unroll
    for (int nt = 0; nt < 8; ++nt) {
      size_t ntg = (size_t)chunk * 8 + nt;
      const v16h b = *(const v16h*)(Bpk + ((ntg * numKt + kk) << 9) + lane * 16);
      acc[nt] = wmma_f16(a, b, acc[nt]);
    }
  }
  int nin = lane & 15;
  if (tileStore) {
    size_t ntTotal = (size_t)numChunks * 8;
#pragma unroll
    for (int nt = 0; nt < 8; ++nt) {
      int ntg = chunk * 8 + nt;
      float bv = bias[ntg * 16 + nin];
      v8f r;
#pragma unroll
      for (int v = 0; v < 8; ++v) r[v] = acc[nt][v] + bv;
      *(v8f*)(C + ((size_t)mtile * ntTotal + ntg) * 256 + lane * 8) = r;
    }
  } else {
    int mh = (lane >> 4) << 3;
#pragma unroll
    for (int nt = 0; nt < 8; ++nt) {
      int col = (chunk * 8 + nt) * 16 + nin;
      float bv = bias[col];
#pragma unroll
      for (int v = 0; v < 8; ++v) {
        float r = acc[nt][v] + bv;
        if (act) r = tanhf(r);
        C[(size_t)(mtile * 16 + mh + v) * ldc + col] = r;
      }
    }
  }
}

// ---------------------------------------------------------------------------
// LSTM scan: one workgroup, 32 waves.  Wave jt owns j-columns [jt*16,jt*16+16)
// for all 4 gates and both 16-row batch halves.  h lives in LDS as A-operand
// blobs (2 M-tiles x 16 K-tiles x 512 halves = 32KB) so A fetches are single
// v16h ds loads; c in registers.  z arrives as v8f tile blobs streamed in t
// order; hs written as A-blobs (rows t*B+b) for the next GEMM.
// ---------------------------------------------------------------------------
__global__ void __launch_bounds__(1024)
lstm_scan_kernel(const float* __restrict__ zt, const _Float16* __restrict__ Whpk,
                 _Float16* __restrict__ hs) {
  __shared__ _Float16 hbuf[2 * 16 * 512];
  int tid = threadIdx.x;
  int jt = tid >> 5;          // wave id = j-tile, 0..31
  int lane = tid & 31;
  for (int i = tid; i < 2 * 16 * 512; i += 1024) hbuf[i] = (_Float16)0.0f;
  __syncthreads();

  float c[2][8] = {};
  int nin = lane & 15, mh = (lane >> 4) << 3;
  int col = jt * 16 + nin;                      // j index = K of next GEMM
  int ktw = col >> 5, kw = col & 31;
  int abidx[8];
#pragma unroll
  for (int v = 0; v < 8; ++v) abidx[v] = a_blob_idx(mh + v, kw);

  for (int t = 0; t < T_; ++t) {
    v8f acc[4][2];
#pragma unroll
    for (int g = 0; g < 4; ++g)
#pragma unroll
      for (int mt = 0; mt < 2; ++mt)
        acc[g][mt] = *(const v8f*)(zt + ((size_t)(t * 2 + mt) * 128 + g * 32 + jt) * 256 +
                                   lane * 8);
    for (int kk = 0; kk < HP_ / 32; ++kk) {
      v16h a0 = *(const v16h*)(hbuf + (size_t)kk * 512 + lane * 16);
      v16h a1 = *(const v16h*)(hbuf + (size_t)(16 + kk) * 512 + lane * 16);
#pragma unroll
      for (int g = 0; g < 4; ++g) {
        const v16h b =
            *(const v16h*)(Whpk + (((size_t)(g * 32 + jt) * 16 + kk) << 9) + lane * 16);
        acc[g][0] = wmma_f16(a0, b, acc[g][0]);
        acc[g][1] = wmma_f16(a1, b, acc[g][1]);
      }
    }
    float hv[2][8];
#pragma unroll
    for (int mt = 0; mt < 2; ++mt)
#pragma unroll
      for (int v = 0; v < 8; ++v) {
        float ig = sigf(acc[0][mt][v]);
        float fg = sigf(acc[1][mt][v]);
        float gg = tanhf(acc[2][mt][v]);
        float og = sigf(acc[3][mt][v]);
        float cn = fg * c[mt][v] + ig * gg;
        c[mt][v] = cn;
        hv[mt][v] = og * tanhf(cn);
      }
    __syncthreads();                     // everyone done reading old h
#pragma unroll
    for (int mt = 0; mt < 2; ++mt)
#pragma unroll
      for (int v = 0; v < 8; ++v) {
        _Float16 hh = (_Float16)hv[mt][v];
        hbuf[(size_t)(mt * 16 + ktw) * 512 + abidx[v]] = hh;
        hs[(((size_t)(t * 2 + mt) * 16 + ktw) << 9) + abidx[v]] = hh;
      }
    __syncthreads();                     // new h visible to all waves
  }
}

// Tiny final layer: out[b][c] = tanh(x[b,:500] @ W4 + b4), fp32.
__global__ void fc4_kernel(const float* __restrict__ x, const float* __restrict__ W4,
                           const float* __restrict__ b4, float* __restrict__ out) {
  int i = blockIdx.x * blockDim.x + threadIdx.x;
  if (i >= B_ * 10) return;
  int b = i / 10, cix = i % 10;
  float s = b4[cix];
  for (int k = 0; k < H_; ++k) s += x[b * HP_ + k] * W4[k * 10 + cix];
  out[i] = tanhf(s);
}

extern "C" void kernel_launch(void* const* d_in, const int* in_sizes, int n_in,
                              void* d_out, int out_size, void* d_ws, size_t ws_size,
                              hipStream_t stream) {
  (void)in_sizes; (void)n_in; (void)out_size; (void)ws_size;
  const int*   maskp = (const int*)d_in[0];
  const float* embp  = (const float*)d_in[1];
  const float* Wx1p  = (const float*)d_in[2];
  const float* Wh1p  = (const float*)d_in[3];
  const float* b1p_  = (const float*)d_in[4];
  const float* Wx2p  = (const float*)d_in[5];
  const float* Wh2p  = (const float*)d_in[6];
  const float* b2p_  = (const float*)d_in[7];
  const float* W3p_  = (const float*)d_in[8];
  const float* b3p_  = (const float*)d_in[9];
  const float* W4p   = (const float*)d_in[10];
  const float* b4p   = (const float*)d_in[11];
  float* out = (float*)d_out;

  char* ws = (char*)d_ws;
  size_t off = 0;
  auto take = [&](size_t bytes) -> void* {
    void* p = ws + off;
    off += (bytes + 255) & ~(size_t)255;
    return p;
  };
  const int numKtE = (V_ + 31) / 32;   // 313
  _Float16* embpk = (_Float16*)take((size_t)numKtE * 8 * 512 * 2);
  _Float16* wx1pk = (_Float16*)take((size_t)4  * 128 * 512 * 2);
  _Float16* wh1pk = (_Float16*)take((size_t)16 * 128 * 512 * 2);
  _Float16* wx2pk = (_Float16*)take((size_t)16 * 128 * 512 * 2);
  _Float16* wh2pk = (_Float16*)take((size_t)16 * 128 * 512 * 2);
  _Float16* w3pk  = (_Float16*)take((size_t)16 * 32  * 512 * 2);
  float* b1pk = (float*)take(NG_ * 4);
  float* b2pk = (float*)take(NG_ * 4);
  float* b3pk = (float*)take(HP_ * 4);
  _Float16* e16 = (_Float16*)take((size_t)B_ * T_ * E_ * 2);    // A-blobs, rows t*B+b
  float*    zbuf = (float*)take((size_t)B_ * T_ * NG_ * 4);     // z tile blobs
  _Float16* hsb  = (_Float16*)take((size_t)B_ * T_ * HP_ * 2);  // A-blobs, rows t*B+b
  float*    out3 = (float*)take((size_t)B_ * HP_ * 4);

  auto packB = [&](const float* src, int srcK, int srcStride, int J, int Jpad,
                   int numKt, int numNt, _Float16* dst) {
    size_t total = (size_t)numKt * numNt * 512;
    int blocks = (int)((total + 255) / 256);
    pack_b_kernel<<<blocks, 256, 0, stream>>>(src, srcK, srcStride, J, Jpad,
                                              numKt, numNt, dst);
  };
  packB(embp, V_, E_,   E_, E_,  numKtE, 8,   embpk);
  packB(Wx1p, E_, 2000, H_, HP_, 4,      128, wx1pk);
  packB(Wh1p, H_, 2000, H_, HP_, 16,     128, wh1pk);
  packB(Wx2p, H_, 2000, H_, HP_, 16,     128, wx2pk);
  packB(Wh2p, H_, 2000, H_, HP_, 16,     128, wh2pk);
  packB(W3p_, H_, H_,   H_, HP_, 16,     32,  w3pk);
  pack_bias_kernel<<<8, 256, 0, stream>>>(b1p_, 4, H_, HP_, b1pk);
  pack_bias_kernel<<<8, 256, 0, stream>>>(b2p_, 4, H_, HP_, b2pk);
  pack_bias_kernel<<<2, 256, 0, stream>>>(b3p_, 1, H_, HP_, b3pk);

  // e = mask @ emb      (200 blocks x 8 K-split waves; 128MB mask read once)
  embed_gemm_kernel<<<200, 256, 0, stream>>>(maskp, embpk, e16, numKtE);
  // z1 = e @ Wx1 + b1   -> tile blobs  (200 Mtiles x 16 chunks = 3200 waves)
  gemm_f16_kernel<<<400, 256, 0, stream>>>(e16, wx1pk, b1pk,
                                           zbuf, 0, 200, 16, 4, 0, 1);
  // LSTM layer 1 scan
  lstm_scan_kernel<<<1, 1024, 0, stream>>>(zbuf, wh1pk, hsb);
  // z2 = hs @ Wx2 + b2  -> tile blobs
  gemm_f16_kernel<<<400, 256, 0, stream>>>(hsb, wx2pk, b2pk,
                                           zbuf, 0, 200, 16, 16, 0, 1);
  // LSTM layer 2 scan (hsb overwritten; only t=T-1 blobs consumed next)
  lstm_scan_kernel<<<1, 1024, 0, stream>>>(zbuf, wh2pk, hsb);
  // fc3 = tanh(h_last @ W3 + b3): last 2 M-tile blobs of hsb, contiguous
  gemm_f16_kernel<<<1, 256, 0, stream>>>(hsb + (size_t)(T_ - 1) * 2 * 16 * 512,
                                         w3pk, b3pk, out3, HP_, 2, 4, 16, 1, 0);
  // fc4 = tanh(x @ W4 + b4) -> d_out
  fc4_kernel<<<1, 320, 0, stream>>>(out3, W4p, b4p, out);
}
